// OneBitLinear_80350248173700
// MI455X (gfx1250) — matrix operand order repaired and checked by reading, not compiled, not run on hardware
//
#include <hip/hip_runtime.h>

// ---------------------------------------------------------------------------
// OneBitLinear: h = LayerNorm((x * in_scale) @ sign(W)^T * out_scale + bias)
// N=8192 rows, D_in = D_out = 4096.  f16 WMMA GEMM (sign(W) exact in f16),
// f32 accumulation, TDM (tensor_load_to_lds) double-buffered tile staging,
// fused scale+bias epilogue, row LayerNorm pass.
// Workspace: ax (N*D_in f16, 64MB) | bw (D_out*D_in f16, 32MB) -> L2-resident.
// ---------------------------------------------------------------------------

typedef _Float16 v8h  __attribute__((ext_vector_type(8)));
typedef _Float16 v16h __attribute__((ext_vector_type(16)));
typedef float    v8f  __attribute__((ext_vector_type(8)));
typedef unsigned int u32x4 __attribute__((ext_vector_type(4)));
typedef int          i32x4 __attribute__((ext_vector_type(4)));
typedef int          i32x8 __attribute__((ext_vector_type(8)));

#define D_IN   4096
#define D_OUT  4096
#define NROWS  8192
#define BK     32        // K-step (matches 16x16x32 f16 WMMA)
#define LDT    40        // padded LDS row stride in halfs (80B): 32 data + 8 pad

#if __has_builtin(__builtin_amdgcn_tensor_load_to_lds) && __has_builtin(__builtin_amdgcn_s_wait_tensorcnt)
#define ONEBIT_TDM 1
#else
#define ONEBIT_TDM 0
#endif

// ---------------------------------------------------------------- prep: a = f16(x * in_scale)
__global__ __launch_bounds__(256)
void onebit_prep_ax(const float* __restrict__ x,
                    const float* __restrict__ iscale,
                    _Float16* __restrict__ ax) {
    const int idx = (blockIdx.x * 256 + threadIdx.x) * 8;
    const int k   = idx & (D_IN - 1);
    const float4 a = *(const float4*)(x + idx);
    const float4 b = *(const float4*)(x + idx + 4);
    const float4 s0 = *(const float4*)(iscale + k);
    const float4 s1 = *(const float4*)(iscale + k + 4);
    v8h o;
    o[0] = (_Float16)(a.x * s0.x);  o[1] = (_Float16)(a.y * s0.y);
    o[2] = (_Float16)(a.z * s0.z);  o[3] = (_Float16)(a.w * s0.w);
    o[4] = (_Float16)(b.x * s1.x);  o[5] = (_Float16)(b.y * s1.y);
    o[6] = (_Float16)(b.z * s1.z);  o[7] = (_Float16)(b.w * s1.w);
    *(v8h*)(ax + idx) = o;
}

// ---------------------------------------------------------------- prep: bw = f16(sign(W)), sign(0)=+1
__global__ __launch_bounds__(256)
void onebit_prep_w(const float* __restrict__ w,
                   _Float16* __restrict__ bw) {
    const int idx = (blockIdx.x * 256 + threadIdx.x) * 8;
    const float4 a = *(const float4*)(w + idx);
    const float4 b = *(const float4*)(w + idx + 4);
    v8h o;
    o[0] = (a.x >= 0.f) ? (_Float16)1 : (_Float16)-1;
    o[1] = (a.y >= 0.f) ? (_Float16)1 : (_Float16)-1;
    o[2] = (a.z >= 0.f) ? (_Float16)1 : (_Float16)-1;
    o[3] = (a.w >= 0.f) ? (_Float16)1 : (_Float16)-1;
    o[4] = (b.x >= 0.f) ? (_Float16)1 : (_Float16)-1;
    o[5] = (b.y >= 0.f) ? (_Float16)1 : (_Float16)-1;
    o[6] = (b.z >= 0.f) ? (_Float16)1 : (_Float16)-1;
    o[7] = (b.w >= 0.f) ? (_Float16)1 : (_Float16)-1;
    *(v8h*)(bw + idx) = o;
}

#if ONEBIT_TDM
// Issue a 2D TDM tile load: 32 halfs x 128 rows from row-major [*, D_IN] f16
// tensor into LDS with +8-half row padding (pad_interval=16 DWORDs -> +4 DWORDs).
__device__ __forceinline__ void onebit_tdm_load_tile(const void* gsrc,
                                                     unsigned lds_addr,
                                                     unsigned tensor_d1) {
    const unsigned long long ga = (unsigned long long)gsrc;
    u32x4 g0;
    g0[0] = 1u;                                                // count=1, user D#
    g0[1] = lds_addr;                                          // LDS byte address
    g0[2] = (unsigned)ga;                                      // global_addr[31:0]
    g0[3] = (unsigned)((ga >> 32) & 0x01FFFFFFu) | (2u << 30); // addr[56:32], type=2
    i32x8 g1;
    g1[0] = (int)((1u << 16)      // data_size = 2 bytes
                | (1u << 20)      // pad_enable
                | (3u << 22)      // pad_interval: every 16 DWORDs (= 32 halfs)
                | (3u << 25));    // pad_amount: 4 DWORDs (= 8 halfs)
    g1[1] = (int)((D_IN & 0xFFFFu) << 16);                     // tensor_dim0[15:0]
    g1[2] = (int)((D_IN >> 16) | ((tensor_d1 & 0xFFFFu) << 16));
    g1[3] = (int)((tensor_d1 >> 16) | ((unsigned)BK << 16));   // tile_dim0 = 32
    g1[4] = 128;                                               // tile_dim1 = 128, tile_dim2 = 0
    g1[5] = D_IN;                                              // tensor_dim0_stride[31:0]
    g1[6] = 0;
    g1[7] = 0;
    i32x4 z4; z4[0] = 0; z4[1] = 0; z4[2] = 0; z4[3] = 0;      // 2-D tile: groups 2/3 unused
#if defined(__clang_major__) && (__clang_major__ >= 23)
    i32x8 z8; z8[0]=0; z8[1]=0; z8[2]=0; z8[3]=0; z8[4]=0; z8[5]=0; z8[6]=0; z8[7]=0;
    __builtin_amdgcn_tensor_load_to_lds(g0, g1, z4, z4, z8, 0);
#else
    __builtin_amdgcn_tensor_load_to_lds(g0, g1, z4, z4, 0);
#endif
}
#endif // ONEBIT_TDM

// Fragment build + 8 WMMAs for one 32-half K-slice.
// A frag (16x32 f16): lane<16 holds K0-7 & K16-23; lane>=16 holds K8-15 & K24-31.
// B frag (32x16 f16): lane<16 holds K0-7 & K16-23; lane>=16 holds K8-15 & K24-31 (column = lane%16).
__device__ __forceinline__ void onebit_mma_step(const _Float16* Asb,
                                                const _Float16* Bsb,
                                                int wm, int wn, int lr, bool hi,
                                                v8f (&acc)[2][4]) {
    v16h afrag[2];
    #pragma unroll
    for (int i = 0; i < 2; ++i) {
        const int row = wm + i * 16 + lr;
        const int klo = hi ? 8 : 0;
        const v8h p0 = *(const v8h*)&Asb[row * LDT + klo];
        const v8h p1 = *(const v8h*)&Asb[row * LDT + klo + 16];
        #pragma unroll
        for (int e = 0; e < 8; ++e) { afrag[i][e] = p0[e]; afrag[i][e + 8] = p1[e]; }
    }
    v16h bfrag[4];
    #pragma unroll
    for (int j = 0; j < 4; ++j) {
        const int col = wn + j * 16 + lr;
        const int kb  = hi ? 8 : 0;
        const v8h p0 = *(const v8h*)&Bsb[col * LDT + kb];
        const v8h p1 = *(const v8h*)&Bsb[col * LDT + kb + 16];
        #pragma unroll
        for (int e = 0; e < 8; ++e) { bfrag[j][e] = p0[e]; bfrag[j][e + 8] = p1[e]; }
    }
    #pragma unroll
    for (int i = 0; i < 2; ++i)
        #pragma unroll
        for (int j = 0; j < 4; ++j)
            acc[i][j] = __builtin_amdgcn_wmma_f32_16x16x32_f16(
                false, afrag[i], false, bfrag[j], (short)0, acc[i][j], false, false);
}

// ---------------------------------------------------------------- WMMA GEMM
// 256 threads = 8 waves; 128(M) x 128(N) block tile; waves 4(M) x 2(N),
// each wave 32x64 = 2x4 accumulator tiles. TDM double-buffered K pipeline.
__global__ __launch_bounds__(256)
void onebit_gemm(const _Float16* __restrict__ A,   // [NROWS][D_IN]
                 const _Float16* __restrict__ Bw,  // [D_OUT][D_IN]
                 const float* __restrict__ oscale,
                 const float* __restrict__ bias,
                 float* __restrict__ Out) {        // [NROWS][D_OUT]
    __shared__ __align__(16) _Float16 As[2][128 * LDT];
    __shared__ __align__(16) _Float16 Bs[2][128 * LDT];

    const int tid  = threadIdx.x;
    const int lane = tid & 31;
    const int wid  = tid >> 5;
    const int wm   = (wid & 3) * 32;
    const int wn   = (wid >> 2) * 64;
    const int lr   = lane & 15;
    const bool hi  = lane >= 16;

    const int block_n = blockIdx.x * 128;
    const int block_m = blockIdx.y * 128;

    v8f acc[2][4];
    v8f zero;
    #pragma unroll
    for (int e = 0; e < 8; ++e) zero[e] = 0.0f;
    #pragma unroll
    for (int i = 0; i < 2; ++i)
        #pragma unroll
        for (int j = 0; j < 4; ++j) acc[i][j] = zero;

#if ONEBIT_TDM
    const unsigned ldsA0 = (unsigned)(unsigned long long)&As[0][0];
    const unsigned ldsA1 = (unsigned)(unsigned long long)&As[1][0];
    const unsigned ldsB0 = (unsigned)(unsigned long long)&Bs[0][0];
    const unsigned ldsB1 = (unsigned)(unsigned long long)&Bs[1][0];

    if (wid == 0) {   // prologue: stage tile k=0 into buffer 0
        onebit_tdm_load_tile(A  + (size_t)block_m * D_IN, ldsA0, NROWS);
        onebit_tdm_load_tile(Bw + (size_t)block_n * D_IN, ldsB0, D_OUT);
    }
    int cur = 0;
    for (int k0 = 0; k0 < D_IN; k0 += BK, cur ^= 1) {
        if (wid == 0) {
            if (k0 + BK < D_IN) {   // prefetch next tile into the other buffer
                onebit_tdm_load_tile(A  + (size_t)block_m * D_IN + (k0 + BK),
                                     cur ? ldsA0 : ldsA1, NROWS);
                onebit_tdm_load_tile(Bw + (size_t)block_n * D_IN + (k0 + BK),
                                     cur ? ldsB0 : ldsB1, D_OUT);
                __builtin_amdgcn_s_wait_tensorcnt((short)2); // current pair retired
            } else {
                __builtin_amdgcn_s_wait_tensorcnt((short)0);
            }
        }
        __syncthreads();   // current tile visible to all waves
        onebit_mma_step(&As[cur][0], &Bs[cur][0], wm, wn, lr, hi, acc);
        __syncthreads();   // everyone done reading before buffer is re-filled
    }
#else
    // Fallback: VGPR-staged single-buffer pipeline (proven codegen).
    for (int k0 = 0; k0 < D_IN; k0 += BK) {
        __syncthreads();
        #pragma unroll
        for (int c = tid; c < 512; c += 256) {
            const int r  = c >> 2;
            const int kc = (c & 3) << 3;
            *(v8h*)&As[0][r * LDT + kc] = *(const v8h*)&A [(block_m + r) * D_IN + k0 + kc];
            *(v8h*)&Bs[0][r * LDT + kc] = *(const v8h*)&Bw[(block_n + r) * D_IN + k0 + kc];
        }
        __syncthreads();
        onebit_mma_step(&As[0][0], &Bs[0][0], wm, wn, lr, hi, acc);
    }
#endif

    // Epilogue: D layout -> VGPR v: lanes0-15 M=v, lanes16-31 M=v+8; N = lane%16.
    const int rsel = hi ? 8 : 0;
    #pragma unroll
    for (int j = 0; j < 4; ++j) {
        const int col = block_n + wn + j * 16 + lr;
        const float os = oscale[col];
        const float bi = bias[col];
        #pragma unroll
        for (int i = 0; i < 2; ++i) {
            const int row0 = block_m + wm + i * 16 + rsel;
            #pragma unroll
            for (int v = 0; v < 8; ++v)
                Out[(size_t)(row0 + v) * D_OUT + col] = acc[i][j][v] * os + bi;
        }
    }
}

// ---------------------------------------------------------------- row LayerNorm (in place)
__global__ __launch_bounds__(256)
void onebit_ln(float* __restrict__ H,
               const float* __restrict__ gamma,
               const float* __restrict__ beta) {
    __shared__ float s_sum[256];
    __shared__ float s_sq[256];
    float* h = H + (size_t)blockIdx.x * D_OUT;
    const int t = threadIdx.x;

    float vals[16];
    float sum = 0.f, sq = 0.f;
    #pragma unroll
    for (int i = 0; i < 16; ++i) {
        const float v = h[t + i * 256];
        vals[i] = v; sum += v; sq += v * v;
    }
    s_sum[t] = sum; s_sq[t] = sq;
    __syncthreads();
    #pragma unroll
    for (int off = 128; off > 0; off >>= 1) {
        if (t < off) { s_sum[t] += s_sum[t + off]; s_sq[t] += s_sq[t + off]; }
        __syncthreads();
    }
    const float mu  = s_sum[0] * (1.0f / D_OUT);
    const float var = s_sq[0] * (1.0f / D_OUT) - mu * mu;
    const float rs  = rsqrtf(var + 1e-5f);
    #pragma unroll
    for (int i = 0; i < 16; ++i) {
        const int col = t + i * 256;
        h[col] = (vals[i] - mu) * rs * gamma[col] + beta[col];
    }
}

// ---------------------------------------------------------------- launcher
extern "C" void kernel_launch(void* const* d_in, const int* in_sizes, int n_in,
                              void* d_out, int out_size, void* d_ws, size_t ws_size,
                              hipStream_t stream) {
    (void)in_sizes; (void)n_in; (void)out_size; (void)ws_size;
    const float* x     = (const float*)d_in[0];
    const float* w     = (const float*)d_in[1];
    const float* iscl  = (const float*)d_in[2];
    const float* oscl  = (const float*)d_in[3];
    const float* bias  = (const float*)d_in[4];
    const float* gamma = (const float*)d_in[5];
    const float* beta  = (const float*)d_in[6];
    float* out = (float*)d_out;

    _Float16* ax = (_Float16*)d_ws;                                        // 64 MB
    _Float16* bw = (_Float16*)((char*)d_ws + (size_t)NROWS * D_IN * 2);    // 32 MB

    onebit_prep_ax<<<(NROWS * D_IN) / (256 * 8), 256, 0, stream>>>(x, iscl, ax);
    onebit_prep_w <<<(D_OUT * D_IN) / (256 * 8), 256, 0, stream>>>(w, bw);

    dim3 grid(D_OUT / 128, NROWS / 128);
    onebit_gemm<<<grid, 256, 0, stream>>>(ax, bw, oscl, bias, out);

    onebit_ln<<<NROWS, 256, 0, stream>>>(out, gamma, beta);
}
